// DepthNet_51659866636480
// MI455X (gfx1250) — matrix operand order repaired and checked by reading, not compiled, not run on hardware
//
#include <hip/hip_runtime.h>
#include <hip/hip_bf16.h>

typedef __attribute__((ext_vector_type(16))) __bf16 v16bf;
typedef __attribute__((ext_vector_type(8)))  float  v8f;

union Frag { v16bf v; uint4 q[2]; };

static constexpr int NV = 5, C = 32, H = 128, W = 160, D = 48;
static constexpr int HW = H * W;
static constexpr int TH = 4, TW = 16;          // output tile per block
static constexpr int EY = TH + 2, EX = TW + 2; // halo-expanded tile

__device__ __forceinline__ unsigned short f2bf(float x) {
  unsigned u = __float_as_uint(x);
  return (unsigned short)((u + 0x7fffu + ((u >> 16) & 1u)) >> 16); // RNE
}

// ---------------- projection prep (1 thread) ----------------
__device__ __forceinline__ void fuse_proj(const float* pmv, float* P) {
  const float* ext  = pmv;       // proj[:, i, 0]
  const float* intr = pmv + 16;  // proj[:, i, 1]
  for (int i = 0; i < 16; ++i) P[i] = ext[i];
  float t[12];
  for (int r = 0; r < 3; ++r)
    for (int c = 0; c < 4; ++c) {
      float s = 0.f;
      for (int k = 0; k < 3; ++k) s += intr[r*4+k] * ext[k*4+c];
      t[r*4+c] = s;
    }
  for (int i = 0; i < 12; ++i) P[i] = t[i];
}

__device__ __forceinline__ void inv4(const float* m, float* o) {
  float inv[16];
  inv[0]  =  m[5]*m[10]*m[15]-m[5]*m[11]*m[14]-m[9]*m[6]*m[15]+m[9]*m[7]*m[14]+m[13]*m[6]*m[11]-m[13]*m[7]*m[10];
  inv[4]  = -m[4]*m[10]*m[15]+m[4]*m[11]*m[14]+m[8]*m[6]*m[15]-m[8]*m[7]*m[14]-m[12]*m[6]*m[11]+m[12]*m[7]*m[10];
  inv[8]  =  m[4]*m[9]*m[15]-m[4]*m[11]*m[13]-m[8]*m[5]*m[15]+m[8]*m[7]*m[13]+m[12]*m[5]*m[11]-m[12]*m[7]*m[9];
  inv[12] = -m[4]*m[9]*m[14]+m[4]*m[10]*m[13]+m[8]*m[5]*m[14]-m[8]*m[6]*m[13]-m[12]*m[5]*m[10]+m[12]*m[6]*m[9];
  inv[1]  = -m[1]*m[10]*m[15]+m[1]*m[11]*m[14]+m[9]*m[2]*m[15]-m[9]*m[3]*m[14]-m[13]*m[2]*m[11]+m[13]*m[3]*m[10];
  inv[5]  =  m[0]*m[10]*m[15]-m[0]*m[11]*m[14]-m[8]*m[2]*m[15]+m[8]*m[3]*m[14]+m[12]*m[2]*m[11]-m[12]*m[3]*m[10];
  inv[9]  = -m[0]*m[9]*m[15]+m[0]*m[11]*m[13]+m[8]*m[1]*m[15]-m[8]*m[3]*m[13]-m[12]*m[1]*m[11]+m[12]*m[3]*m[9];
  inv[13] =  m[0]*m[9]*m[14]-m[0]*m[10]*m[13]-m[8]*m[1]*m[14]+m[8]*m[2]*m[13]+m[12]*m[1]*m[10]-m[12]*m[2]*m[9];
  inv[2]  =  m[1]*m[6]*m[15]-m[1]*m[7]*m[14]-m[5]*m[2]*m[15]+m[5]*m[3]*m[14]+m[13]*m[2]*m[7]-m[13]*m[3]*m[6];
  inv[6]  = -m[0]*m[6]*m[15]+m[0]*m[7]*m[14]+m[4]*m[2]*m[15]-m[4]*m[3]*m[14]-m[12]*m[2]*m[7]+m[12]*m[3]*m[6];
  inv[10] =  m[0]*m[5]*m[15]-m[0]*m[7]*m[13]-m[4]*m[1]*m[15]+m[4]*m[3]*m[13]+m[12]*m[1]*m[7]-m[12]*m[3]*m[5];
  inv[14] = -m[0]*m[5]*m[14]+m[0]*m[6]*m[13]+m[4]*m[1]*m[14]-m[4]*m[2]*m[13]-m[12]*m[1]*m[6]+m[12]*m[2]*m[5];
  inv[3]  = -m[1]*m[6]*m[11]+m[1]*m[7]*m[10]+m[5]*m[2]*m[11]-m[5]*m[3]*m[10]-m[9]*m[2]*m[7]+m[9]*m[3]*m[6];
  inv[7]  =  m[0]*m[6]*m[11]-m[0]*m[7]*m[10]-m[4]*m[2]*m[11]+m[4]*m[3]*m[10]+m[8]*m[2]*m[7]-m[8]*m[3]*m[6];
  inv[11] = -m[0]*m[5]*m[11]+m[0]*m[7]*m[9]+m[4]*m[1]*m[11]-m[4]*m[3]*m[9]-m[8]*m[1]*m[7]+m[8]*m[3]*m[5];
  inv[15] =  m[0]*m[5]*m[10]-m[0]*m[6]*m[9]-m[4]*m[1]*m[10]+m[4]*m[2]*m[9]+m[8]*m[1]*m[6]-m[8]*m[2]*m[5];
  float det = m[0]*inv[0] + m[1]*inv[4] + m[2]*inv[8] + m[3]*inv[12];
  det = 1.0f / det;
  for (int i = 0; i < 16; ++i) o[i] = inv[i] * det;
}

__global__ void proj_prep_kernel(const float* __restrict__ pm, float* __restrict__ ws) {
  if (threadIdx.x != 0 || blockIdx.x != 0) return;
  float ref[16], rinv[16];
  fuse_proj(pm, ref);
  inv4(ref, rinv);
  for (int v = 1; v < NV; ++v) {
    float S[16], P[16];
    fuse_proj(pm + v * 32, S);
    for (int r = 0; r < 4; ++r)
      for (int c = 0; c < 4; ++c) {
        float s = 0.f;
        for (int k = 0; k < 4; ++k) s += S[r*4+k] * rinv[k*4+c];
        P[r*4+c] = s;
      }
    float* o = ws + (v - 1) * 12;
    for (int r = 0; r < 3; ++r) {
      o[r*3+0] = P[r*4+0]; o[r*3+1] = P[r*4+1]; o[r*3+2] = P[r*4+2];
      o[9 + r] = P[r*4+3];
    }
  }
}

// ---------------- fused warp + variance + WMMA conv + softmax ----------------
__global__ __launch_bounds__(128) void mvs_fused_kernel(
    const float* __restrict__ features, const float* __restrict__ dvals,
    const float* __restrict__ regw, const float* __restrict__ vm,
    float* __restrict__ out)
{
  __shared__ alignas(16) unsigned short Vlds[3][EY][EX][C]; // bf16 variance ring (3 depth slices)
  __shared__ alignas(16) unsigned short Wlds[27][C];        // bf16 conv weights [tap][c]
  __shared__ float costLds[D][TH][TW];
  __shared__ float dv_sh[D];

  const int t  = threadIdx.x;
  const int bx = blockIdx.x, by = blockIdx.y;

  for (int i = t; i < 27 * C; i += 128) {         // regw flat = c*27 + tap
    Wlds[i % 27][i / 27] = f2bf(regw[i]);
  }
  if (t < D) dv_sh[t] = dvals[t];

  const int  yy = t / EX, xx = t % EX;            // writer position in expanded tile
  const bool writer = (t < EY * EX);              // 108 writers
  const int  gy = by * TH + yy - 1, gx = bx * TW + xx - 1;
  const bool inImg = writer && gy >= 0 && gy < H && gx >= 0 && gx < W;

  float rxv[4][3], tvv[4][3];                     // per-src-view rot@[x,y,1] and trans
  #pragma unroll
  for (int v = 0; v < 4; ++v) {
    const float* o = vm + v * 12;
    const float fx = (float)gx, fy = (float)gy;
    #pragma unroll
    for (int r = 0; r < 3; ++r) {
      rxv[v][r] = o[r*3+0] * fx + o[r*3+1] * fy + o[r*3+2];
      tvv[v][r] = o[9 + r];
    }
  }

  __syncthreads();

  auto computeSlice = [&](int dd) {
    if (!writer) return;
    const int slot = ((dd % 3) + 3) % 3;
    unsigned short* dst = &Vlds[slot][yy][xx][0];
    if (dd < 0 || dd >= D || !inImg) {            // conv zero padding
      const uint4 z = make_uint4(0u, 0u, 0u, 0u);
      *(uint4*)(dst) = z; *(uint4*)(dst + 8) = z;
      *(uint4*)(dst + 16) = z; *(uint4*)(dst + 24) = z;
      return;
    }
    const float dep = dv_sh[dd];
    int cidx[4][4]; float cw[4][4];
    #pragma unroll
    for (int v = 0; v < 4; ++v) {
      const float p0 = rxv[v][0] * dep + tvv[v][0];
      const float p1 = rxv[v][1] * dep + tvv[v][1];
      const float p2 = rxv[v][2] * dep + tvv[v][2];
      const float px = p0 / p2, py = p1 / p2;
      const float x0 = floorf(px), y0 = floorf(py);
      const float wx1 = px - x0, wx0 = 1.f - wx1;
      const float wy1 = py - y0, wy0 = 1.f - wy1;
      #pragma unroll
      for (int j = 0; j < 4; ++j) {
        const float xc = x0 + (float)(j & 1);
        const float yc = y0 + (float)(j >> 1);
        const float w  = ((j & 1) ? wx1 : wx0) * ((j >> 1) ? wy1 : wy0);
        const bool val = (xc >= 0.f) && (xc <= (float)(W - 1)) &&
                         (yc >= 0.f) && (yc <= (float)(H - 1));
        const int xi = (int)fminf(fmaxf(xc, 0.f), (float)(W - 1));
        const int yi = (int)fminf(fmaxf(yc, 0.f), (float)(H - 1));
        cidx[v][j] = yi * W + xi;
        cw[v][j]   = val ? w : 0.f;
      }
    }
    const int rbase = gy * W + gx;
    #pragma unroll 4
    for (int c = 0; c < C; ++c) {
      const float rv = features[c * HW + rbase];
      float s = rv, q = rv * rv;
      #pragma unroll
      for (int v = 0; v < 4; ++v) {
        const float* f = features + (v + 1) * C * HW + c * HW;
        const float g = f[cidx[v][0]] * cw[v][0] + f[cidx[v][1]] * cw[v][1]
                      + f[cidx[v][2]] * cw[v][2] + f[cidx[v][3]] * cw[v][3];
        s += g; q += g * g;
      }
      const float var = q * 0.2f - (s * 0.2f) * (s * 0.2f); // 1/N = 0.2
      dst[c] = f2bf(var);
    }
  };

  computeSlice(-1);  // zero slice (depth padding)
  computeSlice(0);

  const int wave = t >> 5, lane = t & 31;
  const int m = lane & 15, half = lane >> 4;

  for (int d = 0; d < D; ++d) {
    computeSlice(d + 1);
    __syncthreads();

    // cost[d, wave, m] = sum over 27 taps of <V(c), W(c)> via 27 accumulating WMMAs
    v8f acc = {};
    const int slots[3] = { (d + 2) % 3, d % 3, (d + 1) % 3 };
    #pragma unroll
    for (int kd = 0; kd < 3; ++kd) {
      const unsigned short (*Vs)[EX][C] = Vlds[slots[kd]];
      #pragma unroll
      for (int kh = 0; kh < 3; ++kh) {
        #pragma unroll
        for (int kw = 0; kw < 3; ++kw) {
          const int tap = (kd * 3 + kh) * 3 + kw;
          const unsigned short* pv = &Vs[wave + kh][m + kw][0];
          Frag a, b;
          // A 16x32 bf16: lane(m,half), k = 8*half + (j&7) + 16*(j>=8)
          a.q[0] = *(const uint4*)(pv + half * 8);
          a.q[1] = *(const uint4*)(pv + 16 + half * 8);
          // B 32x16 bf16 (broadcast over n): k = 8*half + (j&7) + 16*(j>=8)
          const unsigned short* pw = &Wlds[tap][0];
          b.q[0] = *(const uint4*)(pw + half * 8);
          b.q[1] = *(const uint4*)(pw + 16 + half * 8);
          acc = __builtin_amdgcn_wmma_f32_16x16x32_bf16(
              false, a.v, false, b.v, (short)0, acc, false, false);
        }
      }
    }
    if (m == 0) {  // lanes 0 (M=0..7) and 16 (M=8..15) hold the 16 results
      #pragma unroll
      for (int r = 0; r < 8; ++r) costLds[d][wave][half * 8 + r] = acc[r];
    }
    __syncthreads();
  }

  // softmax over D + depth regression + 4-window confidence
  if (t < TH * TW) {
    const int y = t >> 4, x = t & 15;
    float mx = costLds[0][y][x];
    for (int d = 1; d < D; ++d) mx = fmaxf(mx, costLds[d][y][x]);
    float s = 0.f, dnum = 0.f, inum = 0.f;
    for (int d = 0; d < D; ++d) {
      const float e = __expf(costLds[d][y][x] - mx);
      s += e; dnum += e * dv_sh[d]; inum += e * (float)d;
    }
    const float depth = dnum / s;
    int didx = (int)(inum / s);
    didx = didx < 0 ? 0 : (didx > D - 1 ? D - 1 : didx);
    float conf = 0.f;
    #pragma unroll
    for (int k = -1; k <= 2; ++k) {
      const int i = didx + k;
      if (i >= 0 && i < D) conf += __expf(costLds[i][y][x] - mx);
    }
    conf /= s;
    const int og = (by * TH + y) * W + (bx * TW + x);
    out[og]      = depth;
    out[HW + og] = conf;
  }
}

extern "C" void kernel_launch(void* const* d_in, const int* in_sizes, int n_in,
                              void* d_out, int out_size, void* d_ws, size_t ws_size,
                              hipStream_t stream) {
  const float* features = (const float*)d_in[0]; // (5,1,32,128,160)
  const float* pm       = (const float*)d_in[1]; // (1,5,2,4,4)
  const float* dvals    = (const float*)d_in[2]; // (1,48)
  const float* regw     = (const float*)d_in[3]; // (1,32,3,3,3)
  float* ws  = (float*)d_ws;                     // 48 floats: per-view rot(9)+trans(3)
  float* out = (float*)d_out;                    // depth (1,128,160) ++ conf (1,128,160)

  proj_prep_kernel<<<1, 1, 0, stream>>>(pm, ws);
  dim3 grid(W / TW, H / TH);                     // (10, 32)
  mvs_fused_kernel<<<grid, 128, 0, stream>>>(features, dvals, regw, ws, out);
}
